// SRHT_1769526526743
// MI455X (gfx1250) — compile-verified, matrix operand order checked by reading, not run on hardware
//
#include <hip/hip_runtime.h>

// SRHT for MI455X (gfx1250, wave32).
//   out[row][j] = FWHT_4096( x[row] * rad[blk] )[ perm[j] mod 4096 ],  blk = j/4096
// One 256-thread workgroup (8 wave32) per row. Row is staged into LDS once
// (x read once: 128 MB total), both blocks' 4096-pt FWHTs computed in LDS,
// permutation gathered from LDS, output written with coalesced float4 stores
// (256 MB). Pure HBM-bound: ~400 MB @ 23.3 TB/s ~= 17 us floor.
//
// FWHT_4096 = H16 (x) H16 (x) H16: three passes of 16x16x16 matmuls
// D = H16 * X_tile on the v_wmma_f32_16x16x4_f32 unit (4 chained K=4 WMMAs),
// Hadamard operand synthesized per-lane as (-1)^popcount(m&k).

#define D_IN   4096
#define NBLK   2
#define D_FEAT 8192

typedef float v2f  __attribute__((ext_vector_type(2)));
typedef float v8f  __attribute__((ext_vector_type(8)));
typedef _Float16 v16h __attribute__((ext_vector_type(16)));

__device__ __forceinline__ float hsign(int i, int j) {
    return (__popc(i & j) & 1) ? -1.0f : 1.0f;
}

__global__ __launch_bounds__(256, 1) void srht_fwht_kernel(
    const float* __restrict__ x,
    const float* __restrict__ rad,     // [NBLK][D_IN] (K==1)
    const int*   __restrict__ perms,   // [D_FEAT], block b's entries in [b*4096,(b+1)*4096)
    float*       __restrict__ out)     // [N][D_FEAT]
{
    __shared__ float lds[NBLK * D_IN];             // 32 KB, blocks contiguous
    const int tid = threadIdx.x;
    const int row = blockIdx.x;
    const float* xr = x + (size_t)row * D_IN;

    // ---- Stage x*rad into LDS for both blocks (coalesced b128 loads/stores) ----
    #pragma unroll
    for (int p = 0; p < 4; ++p) {
        const int i = p * 1024 + tid * 4;
        const float4 xv = *(const float4*)(xr + i);
        const float4 r0 = *(const float4*)(rad + i);
        const float4 r1 = *(const float4*)(rad + D_IN + i);
        float4 y0, y1;
        y0.x = xv.x * r0.x; y0.y = xv.y * r0.y; y0.z = xv.z * r0.z; y0.w = xv.w * r0.w;
        y1.x = xv.x * r1.x; y1.y = xv.y * r1.y; y1.z = xv.z * r1.z; y1.w = xv.w * r1.w;
        *(float4*)(&lds[i])        = y0;
        *(float4*)(&lds[D_IN + i]) = y1;
    }
    __syncthreads();

    const int lane = tid & 31;
    const int wave = tid >> 5;
    const int n    = lane & 15;   // also the A-matrix row M for this lane
    const int h    = lane >> 4;   // lane half select

#if __has_builtin(__builtin_amdgcn_wmma_f32_16x16x4_f32)
    // A = H16 split into 4 K-chunks of 16x4. ISA layout (32-bit A 16x4):
    //   v0: lanes0-15 K=0, lanes16-31 K=2 ; v1: K=1 / K=3  (per chunk, K += 4k)
    v2f A[4];
    #pragma unroll
    for (int k = 0; k < 4; ++k) {
        const int c0 = 4 * k + 2 * h;
        A[k].x = hsign(n, c0);
        A[k].y = hsign(n, c0 + 1);
    }
#else
    // Fallback: H16 zero-padded to 16x32 f16 (exact +-1), f32 accumulate.
    // ISA layout (16-bit A 16x32): lane-half h covers K = 8h..8h+7 in elems 0..7.
    v16h Ah;
    #pragma unroll
    for (int e = 0; e < 16; ++e)
        Ah[e] = (e < 8) ? (_Float16)hsign(n, 8 * h + e) : (_Float16)0.0f;
#endif

    // ---- Three Hadamard passes over the 16x16x16 cube (idx = a*256+b*16+c) ----
    // pass 0: transform c (row stride 1,   col stride 16, tile base t*256)
    // pass 1: transform b (row stride 16,  col stride 1,  tile base t*256)
    // pass 2: transform a (row stride 256, col stride 1,  tile base t*16)
    const int RS[3] = {1, 16, 256};
    const int CS[3] = {16, 1, 1};
    const int TS[3] = {256, 256, 16};

    #pragma unroll
    for (int s = 0; s < 3; ++s) {
        const int rs = RS[s], cs = CS[s], ts = TS[s];
        // 32 tiles (16 per block); each wave owns 4; uniform bounds => EXEC all-1s.
        #pragma unroll
        for (int gi = 0; gi < 4; ++gi) {
            const int g   = wave + gi * 8;
            const int blk = g >> 4;
            const int t   = g & 15;
            float* base = lds + blk * D_IN + t * ts + n * cs;

            v8f c = {};
#if __has_builtin(__builtin_amdgcn_wmma_f32_16x16x4_f32)
            #pragma unroll
            for (int k = 0; k < 4; ++k) {
                const int r0 = 4 * k + 2 * h;      // B 4x16 assumed to mirror A's
                v2f b;                              // even/odd-K split across lanes
                b.x = base[r0 * rs];
                b.y = base[(r0 + 1) * rs];
                c = __builtin_amdgcn_wmma_f32_16x16x4_f32(
                        false, A[k], false, b, (short)0, c, false, false);
            }
#else
            // B 32x16 f16: lanes0-15 K=0..7,16..23 (elem e -> tile row e), lanes16-31 K=8..15,24..31 (zeros).
            v16h Bv;
            #pragma unroll
            for (int e = 0; e < 16; ++e) {
                const float xv = base[(h ? 0 : e) * rs];
                Bv[e] = h ? (_Float16)0.0f : (_Float16)xv;
            }
            c = __builtin_amdgcn_wmma_f32_16x16x32_f16(
                    false, Ah, false, Bv, (short)0, c, false, false);
#endif
            // D layout (32-bit C/D 16x16): vgpr v, lanes0-15 -> M=v, lanes16-31 -> M=v+8.
            // Writes hit exactly the tile's own address set => in-place is safe.
            #pragma unroll
            for (int v = 0; v < 8; ++v)
                base[(v + 8 * h) * rs] = c[v];
        }
        __syncthreads();
    }

    // ---- Permutation gather from LDS, coalesced float4 global stores ----
    float* orow = out + (size_t)row * D_FEAT;
    #pragma unroll
    for (int p = 0; p < 8; ++p) {
        const int j = p * 1024 + tid * 4;
        const int4 pv = *(const int4*)(perms + j);   // hot in L2 (32 KB)
        float4 o;
        o.x = lds[pv.x];   // block offset already baked into perm values
        o.y = lds[pv.y];
        o.z = lds[pv.z];
        o.w = lds[pv.w];
        *(float4*)(orow + j) = o;
    }
}

extern "C" void kernel_launch(void* const* d_in, const int* in_sizes, int n_in,
                              void* d_out, int out_size, void* d_ws, size_t ws_size,
                              hipStream_t stream) {
    (void)n_in; (void)out_size; (void)d_ws; (void)ws_size;
    const float* x     = (const float*)d_in[0];
    const float* rad   = (const float*)d_in[1];
    const int*   perms = (const int*)d_in[2];
    float*       out   = (float*)d_out;
    const int N = in_sizes[0] / D_IN;   // 8192 rows
    srht_fwht_kernel<<<N, 256, 0, stream>>>(x, rad, perms, out);
}
